// cell_56
// MI455X (gfx1250) — matrix operand order repaired and checked by reading, not compiled, not run on hardware
//
#include <hip/hip_runtime.h>
#include <hip/hip_bf16.h>

typedef __attribute__((ext_vector_type(16))) _Float16 v16h;
typedef __attribute__((ext_vector_type(8)))  float    v8f;
typedef __attribute__((ext_vector_type(4)))  _Float16 h4;
typedef __attribute__((ext_vector_type(2)))  _Float16 h2;

#define B_  2048
#define DX_ 64
#define DH_ 128
#define DA_ 128
#define T_  256

__device__ __forceinline__ float sigm(float x) { return 1.0f / (1.0f + __expf(-x)); }

__device__ __forceinline__ v8f wmma16(v16h a, v16h b, v8f c) {
  // D = A(16x32 f16) * B(32x16 f16) + C(16x16 f32)
  return __builtin_amdgcn_wmma_f32_16x16x32_f16(false, a, false, b, (short)0, c, false, false);
}

// A fragment: 16x32 (MxK) f16 tile from row-major LDS, leading dim ld.
// Lanes 0-15: row m=lane, v0..7 = K 0..15; lanes 16-31: K 16..31.
// Per lane this is one contiguous 32B chunk.
__device__ __forceinline__ v16h load_A16(const _Float16* src, int ld) {
  const int lane = threadIdx.x & 31;
  const int m  = lane & 15;
  const int hi = (lane & 16) ? 16 : 0;
  v16h a;
#pragma unroll
  for (int v = 0; v < 8; ++v) {
    const int k = 2 * v + hi;
    a[2 * v]     = src[m * ld + k];
    a[2 * v + 1] = src[m * ld + k + 1];
  }
  return a;
}

// B fragment: 32x16 (KxN) tile from row-major fp32 weights (converted on the fly).
// Lanes 0-15: col n=lane, K=0..15 over v0..7; lanes 16-31: same cols, K=16..31.
__device__ __forceinline__ v16h load_B32(const float* src, int ld) {
  const int lane = threadIdx.x & 31;
  const int n  = lane & 15;
  const int kh = (lane & 16) ? 16 : 0;
  v16h b;
#pragma unroll
  for (int v = 0; v < 8; ++v) {
    const int k = kh + 2 * v;
    b[2 * v]     = (_Float16)src[k * ld + n];
    b[2 * v + 1] = (_Float16)src[(k + 1) * ld + n];
  }
  return b;
}

// Same as load_B32 but from an f16 LDS copy.
__device__ __forceinline__ v16h load_B16(const _Float16* src, int ld) {
  const int lane = threadIdx.x & 31;
  const int n  = lane & 15;
  const int kh = (lane & 16) ? 16 : 0;
  v16h b;
#pragma unroll
  for (int v = 0; v < 8; ++v) {
    const int k = kh + 2 * v;
    b[2 * v]     = src[k * ld + n];
    b[2 * v + 1] = src[(k + 1) * ld + n];
  }
  return b;
}

// fp32 -> f16 LDS staging, float4 / h4 vectorized (count must be multiple of 4)
__device__ __forceinline__ void stage_f16(_Float16* dst, const float* src, int count, int tid, int nthr) {
  const float4* s4 = (const float4*)src;
  h4* d4 = (h4*)dst;
  for (int i = tid; i < count / 4; i += nthr) {
    const float4 v = s4[i];
    h4 o;
    o[0] = (_Float16)v.x; o[1] = (_Float16)v.y; o[2] = (_Float16)v.z; o[3] = (_Float16)v.w;
    d4[i] = o;
  }
}

// x@Wx + h@Wh + s@Ws with register-cached A fragments.
__device__ __forceinline__ v8f gate_mm(const v16h ax[2], const v16h ah[4], const v16h as[4],
                                       const float* Wx, const float* Wh, const float* Ws, int n0) {
  v8f acc = {};
#pragma unroll
  for (int kk = 0; kk < 2; ++kk)
    acc = wmma16(ax[kk], load_B32(Wx + kk * 32 * DH_ + n0, DH_), acc);
#pragma unroll
  for (int kk = 0; kk < 4; ++kk)
    acc = wmma16(ah[kk], load_B32(Wh + kk * 32 * DH_ + n0, DH_), acc);
#pragma unroll
  for (int kk = 0; kk < 4; ++kk)
    acc = wmma16(as[kk], load_B32(Ws + kk * 32 * DH_ + n0, DH_), acc);
  return acc;
}

// ---------------- Kernel 1: gates, cell update, q, g -----------------------
__global__ __launch_bounds__(256) void gates_kernel(
    const float* __restrict__ x_t, const float* __restrict__ prev_h,
    const float* __restrict__ prev_c, const float* __restrict__ dt,
    const float* W_sx, const float* W_sh, const float* w_st, const float* b_s,
    const float* W_fx, const float* W_fh, const float* W_fs, const float* b_f,
    const float* W_ix, const float* W_ih, const float* W_is, const float* b_i,
    const float* W_Tx, const float* W_Th, const float* W_Ts, const float* b_T,
    const float* W_ex, const float* W_eh, const float* W_es, const float* b_e,
    const float* W_ox, const float* W_oh, const float* W_os, const float* b_o,
    const float* W_aq, const float* W_ak, const float* W_hg,
    float* __restrict__ out_newc, float* __restrict__ h_ws,
    float* __restrict__ q_ws, float* __restrict__ g_ws)
{
  __shared__ _Float16 xf[16 * DX_];
  __shared__ _Float16 hf[16 * DH_];
  __shared__ _Float16 sf[16 * DH_];
  __shared__ _Float16 hnf[16 * DH_];
  __shared__ _Float16 cnf[16 * DH_];

  const int b0  = blockIdx.x * 16;
  const int tid = threadIdx.x;
  stage_f16(xf, x_t + b0 * DX_, 16 * DX_, tid, 256);
  stage_f16(hf, prev_h + b0 * DH_, 16 * DH_, tid, 256);
  __syncthreads();

  const int wave = tid >> 5, lane = tid & 31;
  const int n0  = wave * 16;
  const int nn  = n0 + (lane & 15);
  const int mhi = (lane & 16) ? 8 : 0;

  // cache A fragments (x: 2 K-chunks, prev_h: 4 K-chunks)
  v16h ax[2], ah[4];
#pragma unroll
  for (int kk = 0; kk < 2; ++kk) ax[kk] = load_A16(xf + kk * 32, DX_);
#pragma unroll
  for (int kk = 0; kk < 4; ++kk) ah[kk] = load_A16(hf + kk * 32, DH_);

  // --- s gate (needed by all other gates) ---
  v8f acc = {};
#pragma unroll
  for (int kk = 0; kk < 2; ++kk)
    acc = wmma16(ax[kk], load_B32(W_sx + kk * 32 * DH_ + n0, DH_), acc);
#pragma unroll
  for (int kk = 0; kk < 4; ++kk)
    acc = wmma16(ah[kk], load_B32(W_sh + kk * 32 * DH_ + n0, DH_), acc);

  const float wst = w_st[nn], bs = b_s[nn];
  v8f sv;
#pragma unroll
  for (int v = 0; v < 8; ++v) {
    const int m = v + mhi;
    const float s = tanhf(acc[v] + dt[b0 + m] * wst + bs);
    sv[v] = s;
    sf[m * DH_ + nn] = (_Float16)s;
  }
  __syncthreads();

  v16h as[4];
#pragma unroll
  for (int kk = 0; kk < 4; ++kk) as[kk] = load_A16(sf + kk * 32, DH_);

  // --- f, i, T, e, o gates ---
  v8f af = gate_mm(ax, ah, as, W_fx, W_fh, W_fs, n0);
  v8f ai = gate_mm(ax, ah, as, W_ix, W_ih, W_is, n0);
  v8f aT = gate_mm(ax, ah, as, W_Tx, W_Th, W_Ts, n0);
  v8f ae = gate_mm(ax, ah, as, W_ex, W_eh, W_es, n0);
  v8f ao = gate_mm(ax, ah, as, W_ox, W_oh, W_os, n0);

  const float bf_ = b_f[nn], bi_ = b_i[nn], bT_ = b_T[nn], be_ = b_e[nn], bo_ = b_o[nn];
#pragma unroll
  for (int v = 0; v < 8; ++v) {
    const int m = v + mhi;
    const int gi = (b0 + m) * DH_ + nn;
    const float pc = prev_c[gi];
    const float f  = sigm(af[v] + bf_);
    const float ii = sigm(ai[v] + bi_);
    const float Tg = sigm(aT[v] + bT_);
    const float el = tanhf(ae[v] + be_);
    const float o  = sigm(ao[v] + bo_);
    const float nc = f * pc + ii * el + Tg * sv[v];
    const float h  = o * sigm(nc);
    out_newc[gi] = nc;
    h_ws[gi]     = h;
    hnf[m * DH_ + nn] = (_Float16)h;
    cnf[m * DH_ + nn] = (_Float16)nc;
  }
  __syncthreads();

  // --- q = h@W_aq + new_c@W_ak ; g = sigmoid(x@W_hg) ---
  v8f aq = {};
#pragma unroll
  for (int kk = 0; kk < 4; ++kk)
    aq = wmma16(load_A16(hnf + kk * 32, DH_), load_B32(W_aq + kk * 32 * DA_ + n0, DA_), aq);
#pragma unroll
  for (int kk = 0; kk < 4; ++kk)
    aq = wmma16(load_A16(cnf + kk * 32, DH_), load_B32(W_ak + kk * 32 * DA_ + n0, DA_), aq);

  v8f ag = {};
#pragma unroll
  for (int kk = 0; kk < 2; ++kk)
    ag = wmma16(ax[kk], load_B32(W_hg + kk * 32 * DH_ + n0, DH_), ag);

#pragma unroll
  for (int v = 0; v < 8; ++v) {
    const int m = v + mhi;
    const int gi = (b0 + m) * DH_ + nn;
    q_ws[gi] = aq[v];
    g_ws[gi] = sigm(ag[v]);
  }
}

// ---------------- Kernel 2: additive attention (flash-style, 1 pass) -------
// One block per batch row. store[b] is staged once in LDS (f16) and reused for
// both the keys GEMM (WMMA) and the alpha-weighted ctx reduction -> store is
// read from HBM exactly once per element across the whole pipeline.
__global__ __launch_bounds__(256) void attn_kernel(
    const float* __restrict__ store, const float* __restrict__ W_ah,
    const float* __restrict__ b_a, const float* __restrict__ v_a,
    const float* __restrict__ q_ws, float* __restrict__ ctx_ws)
{
  extern __shared__ char smem[];
  _Float16* st16 = (_Float16*)smem;                 // [T_, DH_] f16 copy of store[b]
  _Float16* wah  = st16 + T_ * DH_;                 // [DH_, DA_] f16 copy of W_ah
  float* se   = (float*)(wah + DH_ * DA_);          // [T_] scores -> alpha
  float* sred = se + T_;                            // [T_] reduction scratch
  float* qb   = sred + T_;                          // [DA_] q[b] + b_a
  float* vsh  = qb + DA_;                           // [DA_] v_a
  float* ctxp = vsh + DA_;                          // [4][DH_] ctx partials

  const int b   = blockIdx.x;
  const int tid = threadIdx.x;
  const float* sb = store + (size_t)b * T_ * DH_;

  stage_f16(st16, sb, T_ * DH_, tid, 256);
  stage_f16(wah, W_ah, DH_ * DA_, tid, 256);
  if (tid < DA_) { qb[tid] = q_ws[b * DA_ + tid] + b_a[tid]; vsh[tid] = v_a[tid]; }
  se[tid] = 0.0f;                                   // accumulated via LDS atomics
  __syncthreads();

  const int wave = tid >> 5, lane = tid & 31;
  const int n0  = wave * 16;                        // wave owns one N(=DA)-tile
  const int nl  = lane & 15;
  const int mhi = (lane & 16) ? 8 : 0;

  // B fragments for this wave's N-tile: loaded from LDS once, reused for all t
  v16h bfrag[4];
#pragma unroll
  for (int kk = 0; kk < 4; ++kk) bfrag[kk] = load_B16(wah + kk * 32 * DA_ + n0, DA_);
  const float q_ = qb[n0 + nl], vv = vsh[n0 + nl];

  for (int tt = 0; tt < T_ / 16; ++tt) {
    const int t0 = tt * 16;
    v8f acc = {};
#pragma unroll
    for (int kk = 0; kk < 4; ++kk)
      acc = wmma16(load_A16(st16 + t0 * DH_ + kk * 32, DH_), bfrag[kk], acc);
    float er[8];
#pragma unroll
    for (int v = 0; v < 8; ++v) er[v] = tanhf(acc[v] + q_) * vv;
    // sum this wave's 16 columns per row (across the 16 lanes of each half-wave)
#pragma unroll
    for (int mask = 1; mask < 16; mask <<= 1)
#pragma unroll
      for (int v = 0; v < 8; ++v) er[v] += __shfl_xor(er[v], mask, 32);
    if (nl == 0) {
#pragma unroll
      for (int v = 0; v < 8; ++v) atomicAdd(&se[t0 + mhi + v], er[v]);
    }
  }
  __syncthreads();

  // softmax over T_=256 (one element per thread)
  const float x = se[tid];
  sred[tid] = x; __syncthreads();
  for (int s = 128; s > 0; s >>= 1) {
    if (tid < s) sred[tid] = fmaxf(sred[tid], sred[tid + s]);
    __syncthreads();
  }
  const float mx = sred[0]; __syncthreads();
  const float ex = __expf(x - mx);
  sred[tid] = ex; __syncthreads();
  for (int s = 128; s > 0; s >>= 1) {
    if (tid < s) sred[tid] += sred[tid + s];
    __syncthreads();
  }
  const float inv = 1.0f / sred[0]; __syncthreads();
  se[tid] = ex * inv;
  __syncthreads();

  // ctx = alpha @ store[b] from the LDS-resident copy; h2 pairs, 4-way t-split
  const int d2 = (tid & 63) * 2;
  const int qd = tid >> 6;
  float p0 = 0.0f, p1 = 0.0f;
  for (int t = qd * 64; t < qd * 64 + 64; ++t) {
    const h2 sv2 = *(const h2*)(st16 + t * DH_ + d2);
    const float al = se[t];
    p0 += al * (float)sv2[0];
    p1 += al * (float)sv2[1];
  }
  ctxp[qd * DH_ + d2]     = p0;
  ctxp[qd * DH_ + d2 + 1] = p1;
  __syncthreads();
  if (tid < DH_)
    ctx_ws[b * DH_ + tid] = ctxp[tid] + ctxp[DH_ + tid] + ctxp[2 * DH_ + tid] + ctxp[3 * DH_ + tid];
}

// ---------------- Kernel 3: gated fusion -> new_h --------------------------
__global__ __launch_bounds__(256) void fuse_kernel(
    const float* __restrict__ h_ws, const float* __restrict__ ctx_ws,
    const float* __restrict__ g_ws, const float* W_hh, const float* W_he,
    const float* b_h, float* __restrict__ out_newh)
{
  __shared__ _Float16 hfl[16 * DH_];
  __shared__ _Float16 cfl[16 * DH_];
  const int b0 = blockIdx.x * 16;
  const int tid = threadIdx.x;
  stage_f16(hfl, h_ws + b0 * DH_, 16 * DH_, tid, 256);
  stage_f16(cfl, ctx_ws + b0 * DH_, 16 * DH_, tid, 256);
  __syncthreads();

  const int wave = tid >> 5, lane = tid & 31;
  const int n0  = wave * 16;
  const int nn  = n0 + (lane & 15);
  const int mhi = (lane & 16) ? 8 : 0;

  v16h ah[4], ac[4];
#pragma unroll
  for (int kk = 0; kk < 4; ++kk) {
    ah[kk] = load_A16(hfl + kk * 32, DH_);
    ac[kk] = load_A16(cfl + kk * 32, DH_);
  }

  v8f acc = {};
#pragma unroll
  for (int kk = 0; kk < 4; ++kk)
    acc = wmma16(ah[kk], load_B32(W_hh + kk * 32 * DH_ + n0, DH_), acc);
#pragma unroll
  for (int kk = 0; kk < 4; ++kk)
    acc = wmma16(ac[kk], load_B32(W_he + kk * 32 * DH_ + n0, DH_), acc);

  const float bh = b_h[nn];
#pragma unroll
  for (int v = 0; v < 8; ++v) {
    const int m = v + mhi;
    const int gi = (b0 + m) * DH_ + nn;
    out_newh[gi] = g_ws[gi] * tanhf(acc[v] + bh);
  }
}

extern "C" void kernel_launch(void* const* d_in, const int* in_sizes, int n_in,
                              void* d_out, int out_size, void* d_ws, size_t ws_size,
                              hipStream_t stream) {
  // inputs in setup_inputs() dict order
  const float* x_t    = (const float*)d_in[0];
  const float* prev_h = (const float*)d_in[1];
  const float* prev_c = (const float*)d_in[2];
  const float* dt     = (const float*)d_in[3];
  const float* store  = (const float*)d_in[4];
  const float* W_sx = (const float*)d_in[5],  *W_sh = (const float*)d_in[6],
             * w_st = (const float*)d_in[7],  *b_s  = (const float*)d_in[8];
  const float* W_fx = (const float*)d_in[9],  *W_fh = (const float*)d_in[10],
             * W_fs = (const float*)d_in[11], *b_f  = (const float*)d_in[12];
  const float* W_ix = (const float*)d_in[13], *W_ih = (const float*)d_in[14],
             * W_is = (const float*)d_in[15], *b_i  = (const float*)d_in[16];
  const float* W_Tx = (const float*)d_in[17], *W_Th = (const float*)d_in[18],
             * W_Ts = (const float*)d_in[19], *b_T  = (const float*)d_in[20];
  const float* W_ex = (const float*)d_in[21], *W_eh = (const float*)d_in[22],
             * W_es = (const float*)d_in[23], *b_e  = (const float*)d_in[24];
  const float* W_ox = (const float*)d_in[25], *W_oh = (const float*)d_in[26],
             * W_os = (const float*)d_in[27], *b_o  = (const float*)d_in[28];
  const float* W_aq = (const float*)d_in[29], *W_ak = (const float*)d_in[30],
             * W_ah = (const float*)d_in[31], *b_a  = (const float*)d_in[32],
             * v_a  = (const float*)d_in[33];
  const float* W_hh = (const float*)d_in[34], *W_he = (const float*)d_in[35],
             * W_hg = (const float*)d_in[36], *b_h  = (const float*)d_in[37];

  float* out_newc = (float*)d_out;                  // [B, DH]
  float* out_newh = out_newc + B_ * DH_;            // [B, DH]

  float* h_ws   = (float*)d_ws;                     // [B, DH]
  float* q_ws   = h_ws + B_ * DH_;                  // [B, DA]
  float* g_ws   = q_ws + B_ * DA_;                  // [B, DH]
  float* ctx_ws = g_ws + B_ * DH_;                  // [B, DH]

  gates_kernel<<<B_ / 16, 256, 0, stream>>>(
      x_t, prev_h, prev_c, dt,
      W_sx, W_sh, w_st, b_s,
      W_fx, W_fh, W_fs, b_f,
      W_ix, W_ih, W_is, b_i,
      W_Tx, W_Th, W_Ts, b_T,
      W_ex, W_eh, W_es, b_e,
      W_ox, W_oh, W_os, b_o,
      W_aq, W_ak, W_hg,
      out_newc, h_ws, q_ws, g_ws);

  const size_t attn_lds = (size_t)(T_ * DH_ + DH_ * DA_) * sizeof(_Float16) +
                          (size_t)(T_ + T_ + DA_ + DA_ + 4 * DH_) * sizeof(float);
  attn_kernel<<<B_, 256, attn_lds, stream>>>(store, W_ah, b_a, v_a, q_ws, ctx_ws);

  fuse_kernel<<<B_ / 16, 256, 0, stream>>>(h_ws, ctx_ws, g_ws, W_hh, W_he, b_h, out_newh);
}
